// MultiHeadAttention_41266045781000
// MI455X (gfx1250) — matrix operand order repaired and checked by reading, not compiled, not run on hardware
//
#include <hip/hip_runtime.h>

typedef __attribute__((ext_vector_type(16))) _Float16 v16h;
typedef __attribute__((ext_vector_type(8)))  _Float16 v8h;
typedef __attribute__((ext_vector_type(4)))  _Float16 v4h;
typedef __attribute__((ext_vector_type(8)))  float    v8f;

union V16H { v16h v; v8h h[2]; };

#define Bv 4
#define Tv 2048
#define Cv 1024
#define Hv 16
#define Dv 64

// ---------------------------------------------------------------------------
// Kernel 1a: fp32 -> f16 conversion of x, vectorized (float4 in, 4xf16 out).
// ---------------------------------------------------------------------------
__global__ void cvt_f32_f16_v4(const float* __restrict__ in, _Float16* __restrict__ out, int n4) {
  int i = blockIdx.x * blockDim.x + threadIdx.x;
  if (i < n4) {
    const float4 f = ((const float4*)in)[i];
    v4h o;
    o[0] = (_Float16)f.x; o[1] = (_Float16)f.y;
    o[2] = (_Float16)f.z; o[3] = (_Float16)f.w;
    ((v4h*)out)[i] = o;
  }
}

// ---------------------------------------------------------------------------
// Kernel 1b: convert + transpose w_attn: [K,N] f32 -> [N,K] f16.
// Tiled through LDS so both global read and write are coalesced.
// ---------------------------------------------------------------------------
__global__ __launch_bounds__(256) void cvt_transpose_w(
    const float* __restrict__ w, _Float16* __restrict__ wT, int K, int N) {
  __shared__ _Float16 tile[32][33];
  const int kb = blockIdx.y * 32;
  const int nb = blockIdx.x * 32;
  const int tx = threadIdx.x;   // 0..31
  const int ty = threadIdx.y;   // 0..7
  #pragma unroll
  for (int i = 0; i < 4; ++i) {
    const int kk = ty + i * 8;
    tile[kk][tx] = (_Float16)w[(size_t)(kb + kk) * N + nb + tx];
  }
  __syncthreads();
  #pragma unroll
  for (int i = 0; i < 4; ++i) {
    const int nn = ty + i * 8;
    wT[(size_t)(nb + nn) * K + kb + tx] = tile[tx][nn];
  }
}

// ---------------------------------------------------------------------------
// Kernel 2: qkv = x @ w_attn  (M=8192, N=3072, K=1024), f16 in/out, f32 acc.
// Block tile 128(M) x 256(N), 256 threads = 8 waves (2 over M x 4 over N),
// wave tile 64x64 = 4x4 WMMA accumulators (16 WMMAs : 16 frag b128 loads).
// B is pre-transposed to [N,K] so both LDS tiles fill with b128 copies.
// Global tile fetch is register double-buffered to overlap with WMMA.
// ---------------------------------------------------------------------------
__global__ __launch_bounds__(256) void qkv_gemm_f16(
    const _Float16* __restrict__ Ag,    // [M,K] row-major
    const _Float16* __restrict__ BTg,   // [N,K] row-major (transposed w)
    _Float16* __restrict__ Cg,          // [M,N]
    int M, int N, int K)
{
  __shared__ __align__(16) _Float16 As[128][40];   // [m][k], padded stride 40
  __shared__ __align__(16) _Float16 Bs[256][40];   // [n][k], padded stride 40

  const int m0   = blockIdx.y * 128;
  const int n0   = blockIdx.x * 256;
  const int tid  = threadIdx.x;
  const int wid  = tid >> 5;
  const int lane = tid & 31;
  const int lrow = lane & 15;
  const int hi   = lane >> 4;     // 0 = lanes 0-15, 1 = lanes 16-31
  const int wm   = wid & 1;       // wave position over M (0..1)
  const int wn   = wid >> 1;      // wave position over N (0..3)

  v8f acc[4][4] = {};
  v8h aReg[2], bReg[4];

  // chunk c -> (row = c>>2, kcol = (c&3)*8); A: 512 chunks, B: 1024 chunks
  auto loadA = [&](int k0, int i) {
    const int c = tid + i * 256;
    return *(const v8h*)&Ag[(size_t)(m0 + (c >> 2)) * K + k0 + (c & 3) * 8];
  };
  auto loadB = [&](int k0, int i) {
    const int c = tid + i * 256;
    return *(const v8h*)&BTg[(size_t)(n0 + (c >> 2)) * K + k0 + (c & 3) * 8];
  };

  #pragma unroll
  for (int i = 0; i < 2; ++i) aReg[i] = loadA(0, i);
  #pragma unroll
  for (int i = 0; i < 4; ++i) bReg[i] = loadB(0, i);

  for (int k0 = 0; k0 < K; k0 += 32) {
    // commit staged tile to LDS
    #pragma unroll
    for (int i = 0; i < 2; ++i) {
      const int c = tid + i * 256;
      *(v8h*)&As[c >> 2][(c & 3) * 8] = aReg[i];
    }
    #pragma unroll
    for (int i = 0; i < 4; ++i) {
      const int c = tid + i * 256;
      *(v8h*)&Bs[c >> 2][(c & 3) * 8] = bReg[i];
    }
    __syncthreads();

    // prefetch next tile into registers; latency overlaps the WMMAs below
    if (k0 + 32 < K) {
      #pragma unroll
      for (int i = 0; i < 2; ++i) aReg[i] = loadA(k0 + 32, i);
      #pragma unroll
      for (int i = 0; i < 4; ++i) bReg[i] = loadB(k0 + 32, i);
    }

    // A fragment (16x32): lanes 0-15 row=lrow K=0..7,16..23 ; lanes 16-31 K=8..15,24..31
    V16H af[4], bf[4];
    #pragma unroll
    for (int mt = 0; mt < 4; ++mt) {
      const int r = wm * 64 + mt * 16 + lrow;
      af[mt].h[0] = *(const v8h*)&As[r][hi * 8];
      af[mt].h[1] = *(const v8h*)&As[r][16 + hi * 8];
    }
    // B fragment (32x16): lanes 0-15 col=lrow K=0..7,16..23 ; lanes 16-31 K=8..15,24..31
    #pragma unroll
    for (int nt = 0; nt < 4; ++nt) {
      const int c = wn * 64 + nt * 16 + lrow;
      bf[nt].h[0] = *(const v8h*)&Bs[c][hi * 8];
      bf[nt].h[1] = *(const v8h*)&Bs[c][16 + hi * 8];
    }
    #pragma unroll
    for (int mt = 0; mt < 4; ++mt)
      #pragma unroll
      for (int nt = 0; nt < 4; ++nt)
        acc[mt][nt] = __builtin_amdgcn_wmma_f32_16x16x32_f16(
            false, af[mt].v, false, bf[nt].v, (short)0, acc[mt][nt], false, false);
    __syncthreads();
  }

  // C/D layout: VGPR r -> row r (lanes 0-15) / row r+8 (lanes 16-31), col = lane%16
  #pragma unroll
  for (int mt = 0; mt < 4; ++mt)
    #pragma unroll
    for (int nt = 0; nt < 4; ++nt)
      #pragma unroll
      for (int r = 0; r < 8; ++r) {
        const int row = m0 + wm * 64 + mt * 16 + r + hi * 8;
        const int col = n0 + wn * 64 + nt * 16 + lrow;
        Cg[(size_t)row * N + col] = (_Float16)acc[mt][nt][r];
      }
}

// ---------------------------------------------------------------------------
// Kernel 3: causal flash attention over f16 qkv ([B,T,3C]), f32 output.
// 128 threads = 4 waves, each owning 16 q-rows (block = 64 rows).
// V tile (32 keys x 64 d) is staged transposed into LDS cooperatively and
// SHARED by all 4 waves; all waves run the block-uniform causal tile range
// (extra tiles are numerically a no-op: p=0, alpha=1). Per 32-key tile:
// 4 WMMAs for S = Q K^T, register online softmax in C/D layout, P transposed
// through per-wave LDS, 4 WMMAs for O += P V. K reads hit the 192MB L2.
// ---------------------------------------------------------------------------
__global__ __launch_bounds__(128) void flash_attn_f16(
    const _Float16* __restrict__ qkv,   // [B,T,3C] f16
    float* __restrict__ out)            // [B,T,C] f32
{
  __shared__ __align__(16) _Float16 Vt[64][40];       // [d][key], shared by block
  __shared__ __align__(16) _Float16 Plds[4][16][40];  // [wave][qrow][key]

  const int bh   = blockIdx.y;
  const int b    = bh / Hv;
  const int h    = bh % Hv;
  const int wid  = threadIdx.x >> 5;
  const int lane = threadIdx.x & 31;
  const int lrow = lane & 15;
  const int hi   = lane >> 4;
  const int qb   = blockIdx.x * 64;
  const int qbase = qb + wid * 16;

  const size_t rstride = (size_t)3 * Cv;              // halves per token row
  const _Float16* Qp = qkv + (size_t)b * Tv * rstride + (size_t)h * Dv;
  const _Float16* Kp = Qp + Cv;
  const _Float16* Vp = Qp + 2 * Cv;

  // Q fragments for d=0..31 and d=32..63 (A layout, held for the whole loop)
  V16H qf[2];
  {
    const _Float16* qr = Qp + (size_t)(qbase + lrow) * rstride;
    qf[0].h[0] = *(const v8h*)(qr + hi * 8);
    qf[0].h[1] = *(const v8h*)(qr + 16 + hi * 8);
    qf[1].h[0] = *(const v8h*)(qr + 32 + hi * 8);
    qf[1].h[1] = *(const v8h*)(qr + 48 + hi * 8);
  }

  v8f oacc[4] = {};
  float m[8], l[8];
  #pragma unroll
  for (int r = 0; r < 8; ++r) { m[r] = -3.0e38f; l[r] = 0.0f; }

  const int ktLast = (qb + 63) >> 5;   // block-uniform causal tile range
  for (int kt = 0; kt <= ktLast; ++kt) {
    const int k0 = kt * 32;

    // ---- S = Q @ K^T : two 16(q)x16(key) tiles, contraction D=64 = 2 x 32 ----
    v8f s[2];
    #pragma unroll
    for (int t = 0; t < 2; ++t) {
      // B fragment of K^T: lane = key col; halves = d 8h..8h+7, 16+8h..16+8h+7
      const _Float16* kr = Kp + (size_t)(k0 + t * 16 + lrow) * rstride;
      V16H kb0, kb1;
      kb0.h[0] = *(const v8h*)(kr + hi * 8);
      kb0.h[1] = *(const v8h*)(kr + 16 + hi * 8);
      kb1.h[0] = *(const v8h*)(kr + 32 + hi * 8);
      kb1.h[1] = *(const v8h*)(kr + 48 + hi * 8);
      v8f z = {};
      z = __builtin_amdgcn_wmma_f32_16x16x32_f16(false, qf[0].v, false, kb0.v, (short)0, z, false, false);
      z = __builtin_amdgcn_wmma_f32_16x16x32_f16(false, qf[1].v, false, kb1.v, (short)0, z, false, false);
      s[t] = z;
    }

    // ---- cooperative shared staging of V tile (32 keys x 64 d), transposed ----
    #pragma unroll
    for (int i = 0; i < 8; ++i) {
      const int idx = i * 128 + threadIdx.x;   // 1024 dwords
      const int key = idx >> 5;
      const int dp  = idx & 31;
      union { unsigned int u; _Float16 f[2]; } pk;
      pk.u = *(const unsigned int*)(Vp + (size_t)(k0 + key) * rstride + 2 * dp);
      Vt[2 * dp][key]     = pk.f[0];
      Vt[2 * dp + 1][key] = pk.f[1];
    }

    // ---- online softmax in C/D register layout ----
    #pragma unroll
    for (int r = 0; r < 8; ++r) {
      const int row = qbase + r + hi * 8;        // absolute q index
      float s0 = s[0][r];
      float s1 = s[1][r];
      const int c0 = k0 + lrow;
      const int c1 = k0 + 16 + lrow;
      s0 = (c0 <= row) ? s0 * 0.125f : -3.0e38f;  // scale = D^-0.5, causal mask
      s1 = (c1 <= row) ? s1 * 0.125f : -3.0e38f;
      float v = fmaxf(s0, s1);                    // row max across 16-lane half
      v = fmaxf(v, __shfl_xor(v, 1));
      v = fmaxf(v, __shfl_xor(v, 2));
      v = fmaxf(v, __shfl_xor(v, 4));
      v = fmaxf(v, __shfl_xor(v, 8));
      const float mnew = fmaxf(m[r], v);
      const float p0 = __expf(s0 - mnew);
      const float p1 = __expf(s1 - mnew);
      float rs = p0 + p1;
      rs += __shfl_xor(rs, 1);
      rs += __shfl_xor(rs, 2);
      rs += __shfl_xor(rs, 4);
      rs += __shfl_xor(rs, 8);
      const float alpha = __expf(m[r] - mnew);
      m[r] = mnew;
      l[r] = l[r] * alpha + rs;
      #pragma unroll
      for (int t = 0; t < 4; ++t) oacc[t][r] *= alpha;
      // transpose P through LDS into A-fragment layout
      Plds[wid][r + hi * 8][lrow]      = (_Float16)p0;
      Plds[wid][r + hi * 8][16 + lrow] = (_Float16)p1;
    }

    __syncthreads();   // Vt fully staged (all waves) + Plds ordered

    // ---- O += P @ V : A = P (16x32, contraction over keys), B = V tile ----
    V16H pa;
    pa.h[0] = *(const v8h*)&Plds[wid][lrow][hi * 8];
    pa.h[1] = *(const v8h*)&Plds[wid][lrow][16 + hi * 8];
    #pragma unroll
    for (int t = 0; t < 4; ++t) {
      V16H vb;
      vb.h[0] = *(const v8h*)&Vt[t * 16 + lrow][hi * 8];
      vb.h[1] = *(const v8h*)&Vt[t * 16 + lrow][16 + hi * 8];
      oacc[t] = __builtin_amdgcn_wmma_f32_16x16x32_f16(
          false, pa.v, false, vb.v, (short)0, oacc[t], false, false);
    }
    __syncthreads();   // protect Vt/Plds before next iteration's staging
  }

  // ---- normalize and store fp32 output: out[b, row, h*64 + d] ----
  float* obase = out + (size_t)b * Tv * Cv + (size_t)h * Dv;
  #pragma unroll
  for (int r = 0; r < 8; ++r) {
    const float inv = (l[r] > 0.0f) ? 1.0f / l[r] : 0.0f;
    const int row = qbase + r + hi * 8;
    float* orow = obase + (size_t)row * Cv;
    #pragma unroll
    for (int t = 0; t < 4; ++t)
      orow[t * 16 + lrow] = oacc[t][r] * inv;
  }
}

// ---------------------------------------------------------------------------
// Launcher. Workspace layout (f16): x16 (16MB) | w16T (6MB) | qkv (48MB).
// ---------------------------------------------------------------------------
extern "C" void kernel_launch(void* const* d_in, const int* in_sizes, int n_in,
                              void* d_out, int out_size, void* d_ws, size_t ws_size,
                              hipStream_t stream) {
  (void)in_sizes; (void)n_in; (void)out_size; (void)ws_size;
  const float* x = (const float*)d_in[0];     // [B,T,C] f32
  const float* w = (const float*)d_in[1];     // [C,3C]  f32
  float* out = (float*)d_out;                 // [B,T,C] f32

  const size_t nx = (size_t)Bv * Tv * Cv;     // 8,388,608
  const size_t nw = (size_t)Cv * 3 * Cv;      // 3,145,728

  _Float16* x16  = (_Float16*)d_ws;
  _Float16* w16T = x16 + nx;                  // [3C, C] = [N, K]
  _Float16* qkv  = w16T + nw;                 // [B,T,3C] f16

  cvt_f32_f16_v4<<<(int)((nx / 4 + 255) / 256), 256, 0, stream>>>(x, x16, (int)(nx / 4));

  dim3 gt((3 * Cv) / 32, Cv / 32);            // (N/32, K/32) = (96, 32)
  cvt_transpose_w<<<gt, dim3(32, 8), 0, stream>>>(w, w16T, Cv, 3 * Cv);

  dim3 gg((3 * Cv) / 256, (Bv * Tv) / 128);   // 12 x 64 blocks
  qkv_gemm_f16<<<gg, 256, 0, stream>>>(x16, w16T, qkv, Bv * Tv, 3 * Cv, Cv);

  dim3 ga(Tv / 64, Bv * Hv);                  // 32 x 64 blocks
  flash_attn_f16<<<ga, 128, 0, stream>>>(qkv, out);
}
